// PhysicsHeteroGNN_57758720196716
// MI455X (gfx1250) — matrix operand order repaired and checked by reading, not compiled, not run on hardware
//
#include <hip/hip_runtime.h>
#include <stdint.h>

// ---------------------------------------------------------------------------
// PhysicsHeteroGNN for MI455X (gfx1250, wave32, WMMA)
//
// Edge phase dominates (roofline): node features + aggregation buffers kept as
// packed f16 -> gather 256 B/edge, scatter via native global_atomic_pk_add_f16
// (64 packed atomics/edge). ~6.6 GB total edge traffic at 23.3 TB/s.
// Dense phase: fused 4-term WMMA GEMM (v_wmma_f32_16x16x32_f16, f32 accum),
// 12 unique A fragments in VGPRs, double-buffered B fragments, bias+ReLU fused.
// ---------------------------------------------------------------------------

typedef __attribute__((ext_vector_type(16))) _Float16 v16h;
typedef __attribute__((ext_vector_type(8)))  float    v8f;

#define HDIM   128
#define HU     64        // uints per feature row (HDIM halfs)
#define IN_P   16
#define IN_D   8
#define OUTD   3
#define NLAYER 2

// -------------------------- small helpers ----------------------------------
__device__ __forceinline__ unsigned pack2_f16(float a, float b) {
  union { _Float16 h[2]; unsigned u; } p;
  p.h[0] = (_Float16)a; p.h[1] = (_Float16)b;
  return p.u;
}

__device__ __forceinline__ unsigned scale_pk2(unsigned p, float s) {
  union { _Float16 h[2]; unsigned u; } in, out;
  in.u = p;
  out.h[0] = (_Float16)((float)in.h[0] * s);
  out.h[1] = (_Float16)((float)in.h[1] * s);
  return out.u;
}

// native CDNA5 packed-f16 atomic add, no return (tracked by STOREcnt)
__device__ __forceinline__ void atomic_pk_add_f16(unsigned* addr, unsigned data) {
  asm volatile("global_atomic_pk_add_f16 %0, %1, off"
               :: "v"(addr), "v"(data)
               : "memory");
}

// -------------------------- zero fill (u32 granularity) --------------------
__global__ void zero_u32_kernel(unsigned* __restrict__ p, long long n) {
  long long i = (long long)blockIdx.x * blockDim.x + threadIdx.x;
  long long stride = (long long)gridDim.x * blockDim.x;
  for (; i < n; i += stride) p[i] = 0u;
}

// ------------------- convert f32 weights -> f16 (once) ---------------------
__global__ void cvt_weights_kernel(const float* __restrict__ wl,
                                   const float* __restrict__ wr,
                                   unsigned short* __restrict__ wl_h,
                                   unsigned short* __restrict__ wr_h, int n) {
  int i = blockIdx.x * blockDim.x + threadIdx.x;
  if (i >= n) return;
  union { _Float16 h; unsigned short u; } a, b;
  a.h = (_Float16)wl[i];
  b.h = (_Float16)wr[i];
  wl_h[i] = a.u;
  wr_h[i] = b.u;
}

// ------------------------------ encoder ------------------------------------
// out[i,h] = f16(relu(sum_k x[i,k]*w[h,k] + b[h])); K = 16 or 8, weights L0-hot
__global__ void encode_kernel(const float* __restrict__ x,
                              const float* __restrict__ w,
                              const float* __restrict__ b,
                              unsigned short* __restrict__ out, int n, int kin) {
  int idx = blockIdx.x * blockDim.x + threadIdx.x;
  if (idx >= n * HDIM) return;
  int i = idx / HDIM, h = idx % HDIM;
  const float* xr = x + (long long)i * kin;
  const float* wr = w + (long long)h * kin;
  float acc = b[h];
  for (int k = 0; k < kin; ++k) acc = fmaf(xr[k], wr[k], acc);
  union { _Float16 hf; unsigned short u; } cv;
  cv.hf = (_Float16)fmaxf(acc, 0.0f);
  out[(long long)i * HDIM + h] = cv.u;
}

// --------------------- edge scatter (mean numerator) -----------------------
// one wave per edge; lane covers 4 f16 features (uint2 = 8B coalesced gather),
// two packed-f16 atomics per lane into the f16 accumulation buffer.
__global__ void scatter_kernel(const unsigned* __restrict__ feat,
                               const int* __restrict__ esrc,
                               const int* __restrict__ edst,
                               unsigned* __restrict__ agg,
                               float* __restrict__ cnt, int ne) {
  int e = blockIdx.x * (blockDim.x >> 5) + (threadIdx.x >> 5);
  int lane = threadIdx.x & 31;
  if (e >= ne) return;
  int s = esrc[e];
  int d = edst[e];
  const uint2 v =
      *reinterpret_cast<const uint2*>(feat + (long long)s * HU + lane * 2);
  unsigned* dp = agg + (long long)d * HU + lane * 2;
  atomic_pk_add_f16(dp + 0, v.x);
  atomic_pk_add_f16(dp + 1, v.y);
  if (lane == 0) atomicAdd(cnt + d, 1.0f);
}

// ------------------- fused 4-term SAGE WMMA GEMM ---------------------------
// out = f16(relu( (b1+b2) + (agg1/cnt1)@Wl1^T + (agg2/cnt2)@Wl2^T
//                 + hself@Wr1^T + hself@Wr2^T ))
// 128 threads = 4 waves; wave owns 16 output rows, loops 8 column tiles.
// A tiles (self, mean1, mean2) staged in LDS (packed f16, mean-div folded in);
// 12 unique A fragments preloaded to VGPRs; B fragments double-buffered.
__global__ void __launch_bounds__(128)
sage_fused_kernel(const unsigned* __restrict__ hself,
                  const unsigned* __restrict__ agg1, const float* __restrict__ cnt1,
                  const unsigned* __restrict__ agg2, const float* __restrict__ cnt2,
                  const unsigned short* __restrict__ Wl1,
                  const unsigned short* __restrict__ Wl2,
                  const unsigned short* __restrict__ Wr1,
                  const unsigned short* __restrict__ Wr2,
                  const float* __restrict__ b1, const float* __restrict__ b2,
                  unsigned short* __restrict__ out, int nrows) {
  // 68-dword rows (64 data + 4 pad) rotate LDS banks by 4 per row -> the 16
  // lanes of a fragment read touch 64 distinct banks.
  __shared__ __align__(16) unsigned lds[3][64][68];

  const int t  = threadIdx.x;
  const int m0 = blockIdx.x * 64;

  // ---- stage three 64x128 packed-f16 tiles into LDS ----
  for (int idx = t; idx < 64 * 32; idx += 128) {
    int r = idx >> 5;        // row in tile
    int seg = idx & 31;      // uint2 segment (4 f16 features)
    int grow = m0 + r;
    uint2 v0 = make_uint2(0u, 0u), v1 = v0, v2 = v0;
    float inv1 = 1.0f, inv2 = 1.0f;
    if (grow < nrows) {
      long long base = (long long)grow * HU + seg * 2;
      v0 = *reinterpret_cast<const uint2*>(hself + base);
      v1 = *reinterpret_cast<const uint2*>(agg1 + base);
      v2 = *reinterpret_cast<const uint2*>(agg2 + base);
      inv1 = 1.0f / fmaxf(cnt1[grow], 1.0f);
      inv2 = 1.0f / fmaxf(cnt2[grow], 1.0f);
    }
    *reinterpret_cast<uint2*>(&lds[0][r][seg * 2]) = v0;  // self: direct copy
    *reinterpret_cast<uint2*>(&lds[1][r][seg * 2]) =
        make_uint2(scale_pk2(v1.x, inv1), scale_pk2(v1.y, inv1));
    *reinterpret_cast<uint2*>(&lds[2][r][seg * 2]) =
        make_uint2(scale_pk2(v2.x, inv2), scale_pk2(v2.y, inv2));
  }
  __syncthreads();

  const int wave = t >> 5;
  const int lane = t & 31;
  const int lrow = lane & 15;
  const int hiK  = (lane & 16) ? 1 : 0;
  const int arow = wave * 16 + lrow;  // A-fragment source row in tile

  // ---- preload the 12 unique A fragments (tiles: mean1, mean2, self) ----
  // A 16x32 f16 layout (ISA 7.12.2): lanes<16 hold K 0-7 & 16-23,
  // lanes>=16 hold K 8-15 & 24-31 (two 16B chunks).
  union AF { v16h h; uint4 u[2]; };
  v16h afrag[12];
  const int asrcs[3] = {1, 2, 0};
#pragma unroll
  for (int s = 0; s < 3; ++s) {
    const unsigned* atile = &lds[asrcs[s]][arow][0];
#pragma unroll
    for (int kk = 0; kk < 4; ++kk) {
      AF f;
      int c0 = kk * 32 + hiK * 8;
      int c1 = kk * 32 + 16 + hiK * 8;
      f.u[0] = *reinterpret_cast<const uint4*>(atile + (c0 >> 1));
      f.u[1] = *reinterpret_cast<const uint4*>(atile + (c1 >> 1));
      afrag[s * 4 + kk] = f.h;
    }
  }

  const unsigned short* Wmat[4] = {Wl1, Wl2, Wr1, Wr2};
  const int aidx[4] = {0, 1, 2, 2};  // terms 2,3 share the self tile

  for (int nt = 0; nt < 8; ++nt) {
    const int ncol = nt * 16 + lrow;
    const float bias = b1[ncol] + b2[ncol];
    v8f acc;
#pragma unroll
    for (int r = 0; r < 8; ++r) acc[r] = bias;

    // B 32x16 f16: lane = column n; lanes<16 hold K 0-7 & 16-23, lanes>=16 K 8-15 & 24-31.
    union BF { v16h h; uint4 u[2]; };
    BF bf[2];
    auto loadB = [&](int step, BF& dst) {
      int term = step >> 2, kk = step & 3;
      const unsigned* wb = reinterpret_cast<const unsigned*>(
          Wmat[term] + ncol * HDIM + kk * 32 + hiK * 8);
      dst.u[0] = *reinterpret_cast<const uint4*>(wb);
      dst.u[1] = *reinterpret_cast<const uint4*>(wb + 8);
    };
    loadB(0, bf[0]);
#pragma unroll
    for (int step = 0; step < 16; ++step) {
      if (step < 15) loadB(step + 1, bf[(step + 1) & 1]);  // prefetch next B
      const int term = step >> 2, kk = step & 3;
      acc = __builtin_amdgcn_wmma_f32_16x16x32_f16(
          false, afrag[aidx[term] * 4 + kk], false, bf[step & 1].h,
          (short)0, acc, false, false);
    }

    // relu + f16 store; C/D layout: row = r + 8*(lane>=16), col = lane&15
#pragma unroll
    for (int r = 0; r < 8; ++r) {
      int grow = m0 + wave * 16 + r + (hiK ? 8 : 0);
      if (grow < nrows) {
        float vv = acc[r];
        union { _Float16 hf; unsigned short u; } cv;
        cv.hf = (_Float16)(vv > 0.0f ? vv : 0.0f);
        out[(long long)grow * HDIM + ncol] = cv.u;
      }
    }
  }
}

// ------------------------------ decoder ------------------------------------
__global__ void decode_kernel(const unsigned* __restrict__ h,
                              const float* __restrict__ w,
                              const float* __restrict__ b,
                              float* __restrict__ out, int n) {
  int i = blockIdx.x * blockDim.x + threadIdx.x;
  if (i >= n) return;
  float a0 = b[0], a1 = b[1], a2 = b[2];
  const unsigned* hr = h + (long long)i * HU;
  for (int k = 0; k < HDIM; k += 4) {
    union { _Float16 hf[4]; uint2 u; } c;
    c.u = *reinterpret_cast<const uint2*>(hr + (k >> 1));
    float4 v = make_float4((float)c.hf[0], (float)c.hf[1],
                           (float)c.hf[2], (float)c.hf[3]);
    float4 w0 = *reinterpret_cast<const float4*>(w + 0 * HDIM + k);
    float4 w1 = *reinterpret_cast<const float4*>(w + 1 * HDIM + k);
    float4 w2 = *reinterpret_cast<const float4*>(w + 2 * HDIM + k);
    a0 += v.x * w0.x + v.y * w0.y + v.z * w0.z + v.w * w0.w;
    a1 += v.x * w1.x + v.y * w1.y + v.z * w1.z + v.w * w1.w;
    a2 += v.x * w2.x + v.y * w2.y + v.z * w2.z + v.w * w2.w;
  }
  out[(long long)i * 3 + 0] = a0;
  out[(long long)i * 3 + 1] = a1;
  out[(long long)i * 3 + 2] = a2;
}

// ---------------------------------------------------------------------------
extern "C" void kernel_launch(void* const* d_in, const int* in_sizes, int n_in,
                              void* d_out, int out_size, void* d_ws,
                              size_t ws_size, hipStream_t stream) {
  const int NP = in_sizes[0] / IN_P;  // 100000
  const int ND = in_sizes[1] / IN_D;  // 100000
  const int E  = in_sizes[2] / 2;     // 1600000

  const float* x_p     = (const float*)d_in[0];
  const float* x_d     = (const float*)d_in[1];
  const int*   ei_pp   = (const int*)d_in[2];
  const int*   ei_dd   = (const int*)d_in[3];
  const int*   ei_p2d  = (const int*)d_in[4];
  const float* enc_p_w = (const float*)d_in[5];
  const float* enc_p_b = (const float*)d_in[6];
  const float* enc_d_w = (const float*)d_in[7];
  const float* enc_d_b = (const float*)d_in[8];
  const float* lin_l_w = (const float*)d_in[9];
  const float* lin_l_b = (const float*)d_in[10];
  const float* lin_r_w = (const float*)d_in[11];
  const float* dec_p_w = (const float*)d_in[12];
  const float* dec_p_b = (const float*)d_in[13];
  const float* dec_d_w = (const float*)d_in[14];
  const float* dec_d_b = (const float*)d_in[15];

  // ---- workspace carve ----
  char* ws = (char*)d_ws;
  size_t off = 0;
  auto carve = [&](size_t bytes) -> char* {
    char* p = ws + off;
    off += (bytes + 255) & ~size_t(255);
    return p;
  };
  const size_t nhB = (size_t)NP * HDIM * sizeof(unsigned short);  // f16 rows
  unsigned short* hpA = (unsigned short*)carve(nhB);
  unsigned short* hpB = (unsigned short*)carve(nhB);
  unsigned short* hdA = (unsigned short*)carve(nhB);
  unsigned short* hdB = (unsigned short*)carve(nhB);
  unsigned* aggA = (unsigned*)carve(nhB);
  unsigned* aggB = (unsigned*)carve(nhB);
  float* cntA = (float*)carve((size_t)NP * sizeof(float));
  float* cntB = (float*)carve((size_t)NP * sizeof(float));
  const int NW = NLAYER * 4 * HDIM * HDIM;  // 131072 per weight tensor
  unsigned short* WlH = (unsigned short*)carve((size_t)NW * 2);
  unsigned short* WrH = (unsigned short*)carve((size_t)NW * 2);

  const long long nh  = (long long)NP * HDIM;  // elements
  const long long nhu = (long long)NP * HU;    // uints in an f16 feature array

  // ---- weights -> f16 (once per call; deterministic) ----
  cvt_weights_kernel<<<(NW + 255) / 256, 256, 0, stream>>>(lin_l_w, lin_r_w,
                                                           WlH, WrH, NW);

  // ---- encoders ----
  encode_kernel<<<(int)((nh + 255) / 256), 256, 0, stream>>>(
      x_p, enc_p_w, enc_p_b, hpA, NP, IN_P);
  encode_kernel<<<(int)((nh + 255) / 256), 256, 0, stream>>>(
      x_d, enc_d_w, enc_d_b, hdA, ND, IN_D);

  unsigned short* hp = hpA; unsigned short* hpn = hpB;
  unsigned short* hd = hdA; unsigned short* hdn = hdB;

  const int sBlocks = (E + 7) / 8;            // 8 edges / 256-thread block
  const int gBlocksP = (NP + 63) / 64;
  const int gBlocksD = (ND + 63) / 64;
  const size_t HH = (size_t)HDIM * HDIM;

  for (int l = 0; l < NLAYER; ++l) {
    // ===== primal side: pp (Wl0/Wr0/b0) + d2p (Wl3/Wr3/b3) =====
    zero_u32_kernel<<<1024, 256, 0, stream>>>(aggA, nhu);
    zero_u32_kernel<<<1024, 256, 0, stream>>>(aggB, nhu);
    zero_u32_kernel<<<64, 256, 0, stream>>>((unsigned*)cntA, NP);
    zero_u32_kernel<<<64, 256, 0, stream>>>((unsigned*)cntB, NP);
    scatter_kernel<<<sBlocks, 256, 0, stream>>>((const unsigned*)hp, ei_pp,
                                                ei_pp + E, aggA, cntA, E);
    // d2p = reversed p2d: src = dual row (ei_p2d[1]), dst = primal row
    scatter_kernel<<<sBlocks, 256, 0, stream>>>((const unsigned*)hd, ei_p2d + E,
                                                ei_p2d, aggB, cntB, E);
    sage_fused_kernel<<<gBlocksP, 128, 0, stream>>>(
        (const unsigned*)hp, aggA, cntA, aggB, cntB,
        WlH + (l * 4 + 0) * HH, WlH + (l * 4 + 3) * HH,
        WrH + (l * 4 + 0) * HH, WrH + (l * 4 + 3) * HH,
        lin_l_b + (l * 4 + 0) * HDIM, lin_l_b + (l * 4 + 3) * HDIM, hpn, NP);

    // ===== dual side: dd (Wl1/Wr1/b1) + p2d (Wl2/Wr2/b2) =====
    zero_u32_kernel<<<1024, 256, 0, stream>>>(aggA, nhu);
    zero_u32_kernel<<<1024, 256, 0, stream>>>(aggB, nhu);
    zero_u32_kernel<<<64, 256, 0, stream>>>((unsigned*)cntA, ND);
    zero_u32_kernel<<<64, 256, 0, stream>>>((unsigned*)cntB, ND);
    scatter_kernel<<<sBlocks, 256, 0, stream>>>((const unsigned*)hd, ei_dd,
                                                ei_dd + E, aggA, cntA, E);
    scatter_kernel<<<sBlocks, 256, 0, stream>>>((const unsigned*)hp, ei_p2d,
                                                ei_p2d + E, aggB, cntB, E);
    sage_fused_kernel<<<gBlocksD, 128, 0, stream>>>(
        (const unsigned*)hd, aggA, cntA, aggB, cntB,
        WlH + (l * 4 + 1) * HH, WlH + (l * 4 + 2) * HH,
        WrH + (l * 4 + 1) * HH, WrH + (l * 4 + 2) * HH,
        lin_l_b + (l * 4 + 1) * HDIM, lin_l_b + (l * 4 + 2) * HDIM, hdn, ND);

    // swap double buffers
    unsigned short* tp = hp; hp = hpn; hpn = tp;
    unsigned short* td = hd; hd = hdn; hdn = td;
  }

  // ---- decoders (out = [out_primal | out_dual], f32) ----
  float* outf = (float*)d_out;
  decode_kernel<<<(NP + 255) / 256, 256, 0, stream>>>((const unsigned*)hp,
                                                      dec_p_w, dec_p_b, outf, NP);
  decode_kernel<<<(ND + 255) / 256, 256, 0, stream>>>(
      (const unsigned*)hd, dec_d_w, dec_d_b, outf + (size_t)NP * 3, ND);
}
